// Contrast_18957985644809
// MI455X (gfx1250) — matrix operand order repaired and checked by reading, not compiled, not run on hardware
//
#include <hip/hip_runtime.h>
#include <hip/hip_bf16.h>
#include <math.h>

// ---------------------------------------------------------------------------
// NT-Xent contrastive loss, CDNA5 (gfx1250, wave32, WMMA).
//   B = 4096, D = 256, TEMP = 0.5  ->  2B = 8192 rows, scale = 1/TEMP = 2.
//   loss = (1/2B) * sum_k [ log( sum_{j != k} exp(2*sim_kj) ) - 2*pos_k ]
//   sim_kk == 1 exactly (unit rows) -> diagonal handled by subtracting exp(2).
//   Positives computed exactly in fp32; Gram matrix via v_wmma_f32_16x16x32_f16
//   with double-buffered LDS B supertiles staged through the CDNA5 async
//   memory->LDS path (GLOBAL_LOAD_ASYNC_TO_LDS_B128, ASYNCcnt).
// ---------------------------------------------------------------------------

#define NB   4096
#define ND   256
#define N2   8192                   // 2*B rows of Z
#define EXP2C 7.3890560989306495f   // exp(2.0)

#define LROW 264                    // padded LDS row stride in f16 (528 B)
#define NST  128                    // supertiles per column-half (32 cols each)

typedef __attribute__((ext_vector_type(16))) _Float16 v16h;
typedef __attribute__((ext_vector_type(8)))  _Float16 v8h;
typedef __attribute__((ext_vector_type(8)))  float    v8f;
typedef __attribute__((ext_vector_type(4)))  int      v4i;

#if defined(__gfx1250__) && __has_builtin(__builtin_amdgcn_global_load_async_to_lds_b128)
#define USE_ASYNC_LDS 1
typedef __attribute__((address_space(1))) v4i gv4i;   // global int4
typedef __attribute__((address_space(3))) v4i lv4i;   // LDS int4
#define ASYNC_B128(gp, lp) \
  __builtin_amdgcn_global_load_async_to_lds_b128((gv4i*)(gp), (lv4i*)(lp), 0, 0)
#if __has_builtin(__builtin_amdgcn_s_wait_asynccnt)
#define WAIT_ASYNC0() __builtin_amdgcn_s_wait_asynccnt(0)
#else
#define WAIT_ASYNC0() asm volatile("s_wait_asynccnt 0" ::: "memory")
#endif
#else
#define USE_ASYNC_LDS 0
#endif

// ---------------------------------------------------------------------------
// Kernel 1: L2-normalize rows of [x_i ; x_j] into f16 Z (row-major 8192x256).
// ---------------------------------------------------------------------------
__global__ __launch_bounds__(256) void k_norm(const float* __restrict__ xi,
                                              const float* __restrict__ xj,
                                              _Float16* __restrict__ zh) {
  const int wave = threadIdx.x >> 5;
  const int lane = threadIdx.x & 31;
  const int row  = blockIdx.x * 8 + wave;            // 0..8191
  const float* src = (row < NB) ? (xi + (size_t)row * ND)
                                : (xj + (size_t)(row - NB) * ND);
  const float4 a = *(const float4*)(src + lane * 8);
  const float4 b = *(const float4*)(src + lane * 8 + 4);
  float ss = a.x*a.x + a.y*a.y + a.z*a.z + a.w*a.w
           + b.x*b.x + b.y*b.y + b.z*b.z + b.w*b.w;
  #pragma unroll
  for (int m = 1; m < 32; m <<= 1) ss += __shfl_xor(ss, m, 32);
  const float inv = 1.0f / fmaxf(sqrtf(ss), 1e-12f);

  v8h o;
  o[0] = (_Float16)(a.x * inv); o[1] = (_Float16)(a.y * inv);
  o[2] = (_Float16)(a.z * inv); o[3] = (_Float16)(a.w * inv);
  o[4] = (_Float16)(b.x * inv); o[5] = (_Float16)(b.y * inv);
  o[6] = (_Float16)(b.z * inv); o[7] = (_Float16)(b.w * inv);
  *(v8h*)(zh + (size_t)row * ND + lane * 8) = o;
}

// ---------------------------------------------------------------------------
// Kernel 2: exact fp32 positive-pair cosine sims. pos[i] = pos[i+B].
// ---------------------------------------------------------------------------
__global__ __launch_bounds__(256) void k_pos(const float* __restrict__ xi,
                                             const float* __restrict__ xj,
                                             float* __restrict__ pos) {
  const int wave = threadIdx.x >> 5;
  const int lane = threadIdx.x & 31;
  const int i = blockIdx.x * 8 + wave;               // 0..4095
  const float* a = xi + (size_t)i * ND;
  const float* b = xj + (size_t)i * ND;
  const float4 a0 = *(const float4*)(a + lane * 8);
  const float4 a1 = *(const float4*)(a + lane * 8 + 4);
  const float4 b0 = *(const float4*)(b + lane * 8);
  const float4 b1 = *(const float4*)(b + lane * 8 + 4);
  float dot = a0.x*b0.x + a0.y*b0.y + a0.z*b0.z + a0.w*b0.w
            + a1.x*b1.x + a1.y*b1.y + a1.z*b1.z + a1.w*b1.w;
  float na  = a0.x*a0.x + a0.y*a0.y + a0.z*a0.z + a0.w*a0.w
            + a1.x*a1.x + a1.y*a1.y + a1.z*a1.z + a1.w*a1.w;
  float nb  = b0.x*b0.x + b0.y*b0.y + b0.z*b0.z + b0.w*b0.w
            + b1.x*b1.x + b1.y*b1.y + b1.z*b1.z + b1.w*b1.w;
  #pragma unroll
  for (int m = 1; m < 32; m <<= 1) {
    dot += __shfl_xor(dot, m, 32);
    na  += __shfl_xor(na,  m, 32);
    nb  += __shfl_xor(nb,  m, 32);
  }
  if (lane == 0) {
    const float p = dot / (fmaxf(sqrtf(na), 1e-12f) * fmaxf(sqrtf(nb), 1e-12f));
    pos[i]      = p;
    pos[i + NB] = p;
  }
}

// Kernel 3: zero the row-sum accumulator (atomicAdd target).
__global__ __launch_bounds__(256) void k_zero(float* __restrict__ S) {
  S[blockIdx.x * 256 + threadIdx.x] = 0.0f;
}

// ---------------------------------------------------------------------------
// Kernel 4: Gram-matrix row sums of exp(2*sim), fused via WMMA.
//
// Grid (64, 2) x 256 threads (8 waves). Block owns 128 sim-rows
// (wave w -> 16-row subtile) and one column half (128 supertiles of 32 cols).
// B supertiles staged in double-buffered LDS, padded stride 528 B.
//
// Fragment layouts (CDNA5 ISA):
//   A (16x32 f16): lane m=lane&15 -> row i0+m, half h=lane>>4:
//     VGPR 0-7: K = k0 + h*16 + {0..15} contiguous (same order as B)
//   B (32x16 f16) column-major per lane: lane n -> column j0+n,
//     K = k0 + h*16 + {0..15} contiguous.
//   C/D (16x16 f32): VGPR r, half h -> row r + 8*h, col = lane&15.
// ---------------------------------------------------------------------------
__global__ __launch_bounds__(256) void k_gram(const _Float16* __restrict__ zh,
                                              float* __restrict__ S) {
  __shared__ _Float16 lds[2][32][LROW];              // 2 x 16.5 KB

  const int tid   = threadIdx.x;
  const int wave  = tid >> 5;
  const int lane  = tid & 31;
  const int m     = lane & 15;
  const int h     = lane >> 4;
  const int i0    = blockIdx.x * 128 + wave * 16;    // A subtile base row
  const int jbase = blockIdx.y * NST;                // supertile index base

  // Preload all 8 K-slices of the A fragment (reused for 128 supertiles).
  v16h A[8];
  const _Float16* arow = zh + (size_t)(i0 + m) * ND;
  #pragma unroll
  for (int s = 0; s < 8; ++s) {
    const int k0 = s * 32;
    v8h lo = *(const v8h*)(arow + k0 + h * 16);
    v8h hi = *(const v8h*)(arow + k0 + h * 16 + 8);
    A[s] = __builtin_shufflevector(lo, hi, 0,1,2,3,4,5,6,7,8,9,10,11,12,13,14,15);
  }

  // Staging map: thread -> (row r = tid/8, 64-byte chunk c = tid%8).
  const int sr = tid >> 3;
  const int sc = (tid & 7) * 32;                     // f16 offset within row

  float sums[8] = {0.f,0.f,0.f,0.f,0.f,0.f,0.f,0.f};

#if USE_ASYNC_LDS
  // -------- async memory->LDS staging (ASYNCcnt), no VGPR round-trip -------
  {
    const _Float16* g = zh + (size_t)(jbase * 32 + sr) * ND + sc;
    _Float16* l = &lds[0][sr][sc];
    #pragma unroll
    for (int k = 0; k < 4; ++k) ASYNC_B128(g + k * 8, l + k * 8);
  }
  for (int st = 0; st < NST; ++st) {                 // uniform across lanes
    const int cur = st & 1;
    WAIT_ASYNC0();                                   // own staging done
    __syncthreads();                                 // everyone's staging done
    if (st + 1 < NST) {                              // launch next copy; it
      const _Float16* g =                            // runs behind the WMMAs
          zh + (size_t)((jbase + st + 1) * 32 + sr) * ND + sc;
      _Float16* l = &lds[cur ^ 1][sr][sc];
      #pragma unroll
      for (int k = 0; k < 4; ++k) ASYNC_B128(g + k * 8, l + k * 8);
    }
#else
  // -------- register staging; next-tile loads issued BEFORE the WMMAs ------
  v8h stg[4];
  {
    const _Float16* g = zh + (size_t)(jbase * 32 + sr) * ND + sc;
    #pragma unroll
    for (int k = 0; k < 4; ++k) stg[k] = *(const v8h*)(g + k * 8);
  }
  for (int st = 0; st < NST; ++st) {                 // uniform across lanes
    const int cur = st & 1;
    #pragma unroll
    for (int k = 0; k < 4; ++k)                      // waits on loadcnt here,
      *(v8h*)&lds[cur][sr][sc + k * 8] = stg[k];     // after a full WMMA body
    __syncthreads();
    if (st + 1 < NST) {                              // in flight during WMMAs
      const _Float16* g =
          zh + (size_t)((jbase + st + 1) * 32 + sr) * ND + sc;
      #pragma unroll
      for (int k = 0; k < 4; ++k) stg[k] = *(const v8h*)(g + k * 8);
    }
#endif

    // Consume current supertile: 2 column tiles of 16 (4 WMMA chains total).
    #pragma unroll
    for (int t = 0; t < 2; ++t) {
      const _Float16* brow = &lds[cur][t * 16 + m][h * 16];
      v8f c0 = {0.f,0.f,0.f,0.f,0.f,0.f,0.f,0.f};
      v8f c1 = {0.f,0.f,0.f,0.f,0.f,0.f,0.f,0.f};
      #pragma unroll
      for (int s = 0; s < 8; s += 2) {               // two independent chains
        v8h l0 = *(const v8h*)(brow + s * 32);
        v8h h0 = *(const v8h*)(brow + s * 32 + 8);
        v8h l1 = *(const v8h*)(brow + s * 32 + 32);
        v8h h1 = *(const v8h*)(brow + s * 32 + 40);
        v16h b0 = __builtin_shufflevector(l0, h0, 0,1,2,3,4,5,6,7,8,9,10,11,12,13,14,15);
        v16h b1 = __builtin_shufflevector(l1, h1, 0,1,2,3,4,5,6,7,8,9,10,11,12,13,14,15);
        c0 = __builtin_amdgcn_wmma_f32_16x16x32_f16(false, A[s],     false, b0,
                                                    (short)0, c0, false, false);
        c1 = __builtin_amdgcn_wmma_f32_16x16x32_f16(false, A[s + 1], false, b1,
                                                    (short)0, c1, false, false);
      }
      #pragma unroll
      for (int r = 0; r < 8; ++r) sums[r] += __expf(2.0f * (c0[r] + c1[r]));
    }

#if USE_ASYNC_LDS
  }                                                  // barrier at next top
#else
    __syncthreads();                                 // buffer handoff
  }
#endif

  // Reduce across the 16 lanes of each wave half (columns of the tile).
  #pragma unroll
  for (int msk = 1; msk < 16; msk <<= 1) {
    #pragma unroll
    for (int r = 0; r < 8; ++r) sums[r] += __shfl_xor(sums[r], msk, 32);
  }
  if (m == 0) {
    #pragma unroll
    for (int r = 0; r < 8; ++r)
      atomicAdd(&S[i0 + 8 * h + r], sums[r]);        // 2 column-halves per row
  }
}

// ---------------------------------------------------------------------------
// Kernel 5: loss = (1/8192) * sum_k [ log(S_k - e^2) - 2*pos_k ].
// ---------------------------------------------------------------------------
__global__ __launch_bounds__(256) void k_loss(const float* __restrict__ S,
                                              const float* __restrict__ pos,
                                              float* __restrict__ out) {
  float acc = 0.0f;
  for (int k = threadIdx.x; k < N2; k += 256)
    acc += logf(fmaxf(S[k] - EXP2C, 1e-30f)) - 2.0f * pos[k];

  __shared__ float red[8];
  #pragma unroll
  for (int m = 1; m < 32; m <<= 1) acc += __shfl_xor(acc, m, 32);
  const int wave = threadIdx.x >> 5, lane = threadIdx.x & 31;
  if (lane == 0) red[wave] = acc;
  __syncthreads();
  if (wave == 0) {
    float v = (lane < 8) ? red[lane] : 0.0f;
    #pragma unroll
    for (int m = 1; m < 32; m <<= 1) v += __shfl_xor(v, m, 32);
    if (lane == 0) out[0] = v / (float)N2;
  }
}

// ---------------------------------------------------------------------------
extern "C" void kernel_launch(void* const* d_in, const int* in_sizes, int n_in,
                              void* d_out, int out_size, void* d_ws, size_t ws_size,
                              hipStream_t stream) {
  const float* xi = (const float*)d_in[0];
  const float* xj = (const float*)d_in[1];
  float* out = (float*)d_out;

  // Workspace layout: f16 Z (4 MB) | S rowsums (32 KB) | pos (32 KB)
  _Float16* zh = (_Float16*)d_ws;
  float* S   = (float*)((char*)d_ws + (size_t)N2 * ND * sizeof(_Float16));
  float* pos = S + N2;

  k_norm<<<N2 / 8, 256, 0, stream>>>(xi, xj, zh);
  k_pos <<<NB / 8, 256, 0, stream>>>(xi, xj, pos);
  k_zero<<<N2 / 256, 256, 0, stream>>>(S);
  k_gram<<<dim3(64, 2), 256, 0, stream>>>(zh, S);
  k_loss<<<1, 256, 0, stream>>>(S, pos, out);
}
